// LSTMModel_21492016349996
// MI455X (gfx1250) — compile-verified, matrix operand order checked by reading, not run on hardware
//
#include <hip/hip_runtime.h>
#include <hip/hip_bf16.h>

// ---------------------------------------------------------------------------
// Fused 2-layer LSTM + FC head for gfx1250 (MI455X, wave32, WMMA).
//
// One workgroup = 128 threads = 4 wave32s owns a 16-row batch tile and runs
// the entire T=512 recurrence. All 24 weight WMMA fragments are hoisted into
// registers; the recurrent state A = [h1 | x_t | 1 | h2] lives in LDS.
// Bias and the input projection are fused into the GEMM as extra K columns,
// and x is staged into LDS once so the step loop touches no global memory.
//
// Two barriers per step (down from four): because W2's k=50 row is zero,
// layer-2's kc=0,1 A-fragments ([h1_t | x_{t+1} | 1]) are byte-identical to
// what layer 1 needs at step t+1 — so they are cached in registers across the
// loop back-edge, making layer 1 entirely memory-free. Barrier A publishes
// h2; barrier B publishes h1 + x. Gate transcendentals (V_TANH_F32) are
// pinned into the barrier shadows with data-dependent s_barrier_wait.
// ---------------------------------------------------------------------------

typedef _Float16 v8h  __attribute__((ext_vector_type(8)));
typedef _Float16 v16h __attribute__((ext_vector_type(16)));
typedef float    v8f  __attribute__((ext_vector_type(8)));

#define T_SEQ 512
#define MT    16      // batch rows per workgroup
#define HH1   50
#define HH2   64

// Signal the workgroup barrier once this wave's outstanding LDS ops drained.
#define BAR_SIGNAL() asm volatile("s_wait_dscnt 0\n\ts_barrier_signal -1" ::: "memory")
#define BAR_WAIT()   asm volatile("s_barrier_wait -1" ::: "memory")
// Wait variant carrying fake data deps: forces the scheduler to place the
// wait AFTER the gate math producing (a,b), so VALU fills the barrier shadow.
#define BAR_WAIT_DEP(a, b) \
    asm volatile("s_barrier_wait -1" :: "v"(a), "v"(b) : "memory")

#if __has_builtin(__builtin_amdgcn_tanhf)
__device__ __forceinline__ float tanh_hw(float x) { return __builtin_amdgcn_tanhf(x); }
#else
__device__ __forceinline__ float tanh_hw(float x) {
    return 1.0f - 2.0f / (__expf(2.0f * x) + 1.0f);
}
#endif
__device__ __forceinline__ float sig_hw(float x) {
    return 0.5f * tanh_hw(0.5f * x) + 0.5f;      // sigmoid via V_TANH_F32
}

// Load one 16x32 (f16) A-tile or 32x16 B-tile fragment from LDS in the CDNA5
// WMMA register layout:
//   lanes 0-15 : rows 0-15, VGPR0..3 = K0..7,  VGPR4..7 = K16..23
//   lanes 16-31: rows 0-15, VGPR0..3 = K8..15, VGPR4..7 = K24..31
// (B mirrors A with lane = N column, data stored row-major [n][k].)
__device__ __forceinline__ v16h ld_tile(const _Float16* base, int stride,
                                        int rowoff, int koff, int lane) {
    int r  = rowoff + (lane & 15);
    int hi = (lane >> 4) & 1;
    const _Float16* p = base + r * stride + koff + hi * 8;
    v8h lo = *(const v8h*)(p);
    v8h hv = *(const v8h*)(p + 16);
    return __builtin_shufflevector(lo, hv, 0, 1, 2, 3, 4, 5, 6, 7,
                                           8, 9, 10, 11, 12, 13, 14, 15);
}

__global__ __launch_bounds__(128, 1) void lstm2_wmma_fused(
    const float* __restrict__ x,
    const float* __restrict__ w_ih1, const float* __restrict__ w_hh1,
    const float* __restrict__ b_ih1, const float* __restrict__ b_hh1,
    const float* __restrict__ w_ih2, const float* __restrict__ w_hh2,
    const float* __restrict__ b_ih2, const float* __restrict__ b_hh2,
    const float* __restrict__ fc1_w, const float* __restrict__ fc1_b,
    const float* __restrict__ fc2_w, const float* __restrict__ fc2_b,
    const float* __restrict__ fc3_w, const float* __restrict__ fc3_b,
    float* __restrict__ out) {
    // LDS (f16):
    //   W1s [256][64]      : layer1 weights staging (gate-major rows g*64+n),
    //                        K: 0..49 = w_hh1, 50 = w_ih1, 51 = b_ih1+b_hh1
    //   W2s [256][128]     : layer2 weights staging (rows = gate*64+n),
    //                        K: 0..49 = w_ih2, 51 = b_ih2+b_hh2, 64..127 = w_hh2
    //                        (k = 50 row is ZERO: x never feeds layer 2)
    //   As  [16][128]      : A matrix: 0..49 = h1, 50 = x_t, 51 = 1.0,
    //                        64..127 = h2
    //   x_s [T_SEQ+1][16]  : input tile, transposed, last step zero-padded
    extern __shared__ _Float16 smem[];
    _Float16* W1s = smem;                  // 256*64
    _Float16* W2s = W1s + 256 * 64;        // 256*128
    _Float16* As  = W2s + 256 * 128;       // 16*128
    _Float16* x_s = As + MT * 128;         // (T_SEQ+1)*MT

    const int tid   = threadIdx.x;
    const int lane  = tid & 31;
    const int j     = tid >> 5;        // wave id 0..3 -> hidden cols [16j,16j+16)
    const int n     = lane & 15;
    const int hi    = (lane >> 4) & 1;
    const int col   = 16 * j + n;      // this lane's hidden column
    const int gbase = blockIdx.x * MT;

    // ---- stage padded f16 weights into LDS (bias + input proj fused as K cols)
    for (int i = tid; i < 256 * 64; i += blockDim.x) {
        int gcol = i >> 6, k = i & 63;
        int g = gcol >> 6, nn = gcol & 63;
        float v = 0.0f;
        if (nn < HH1) {
            int r = g * HH1 + nn;
            if (k < HH1)       v = w_hh1[r * HH1 + k];
            else if (k == 50)  v = w_ih1[r];                  // IN == 1
            else if (k == 51)  v = b_ih1[r] + b_hh1[r];
        }
        W1s[i] = (_Float16)v;
    }
    for (int i = tid; i < 256 * 128; i += blockDim.x) {
        int gcol = i >> 7, k = i & 127;
        float v = 0.0f;
        if (k < HH1)           v = w_ih2[gcol * HH1 + k];
        else if (k == 51)      v = b_ih2[gcol] + b_hh2[gcol];
        else if (k >= 64)      v = w_hh2[gcol * HH2 + (k - 64)];
        W2s[i] = (_Float16)v;
    }
    // ---- stage the x tile: coalesced global reads, transposed LDS layout ----
    for (int i = tid; i < MT * T_SEQ; i += blockDim.x) {
        int r = i >> 9;                 // i / T_SEQ
        int t = i & (T_SEQ - 1);
        x_s[t * MT + r] = (_Float16)x[(size_t)(gbase + r) * T_SEQ + t];
    }
    if (tid < MT) x_s[T_SEQ * MT + tid] = (_Float16)0.0f;   // pad step T
    for (int i = tid; i < MT * 128; i += blockDim.x) {
        int k = i & 127;
        As[i] = (_Float16)((k == 51) ? 1.0f : 0.0f);   // constant-1 bias column
    }
    if (tid < MT) As[tid * 128 + 50] = x_s[tid];       // x_0
    __syncthreads();

    // ---- hoist all loop-invariant weight fragments into registers -----------
    v16h w1t[8];                       // [kc][g], layer1, 64 VGPRs
#pragma unroll
    for (int kc = 0; kc < 2; ++kc)
#pragma unroll
        for (int g = 0; g < 4; ++g)
            w1t[kc * 4 + g] = ld_tile(W1s, 64, g * 64 + 16 * j, kc * 32, lane);
    v16h w2t[16];                      // [kc][g], layer2, 128 VGPRs
#pragma unroll
    for (int kc = 0; kc < 4; ++kc)
#pragma unroll
        for (int g = 0; g < 4; ++g)
            w2t[kc * 4 + g] = ld_tile(W2s, 128, g * 64 + 16 * j, kc * 32, lane);

    // Bootstrap the cached A fragments: [h1_{-1}=0 | x_0 | 1] (cols 0..63).
    v16h aA0 = ld_tile(As, 128, 0, 0, lane);
    v16h aA1 = ld_tile(As, 128, 0, 32, lane);

    float c1[8] = {0, 0, 0, 0, 0, 0, 0, 0};
    float c2[8] = {0, 0, 0, 0, 0, 0, 0, 0};
    const v8f vzero = {};

    BAR_SIGNAL();                      // sig A for t = 0 (drains aA loads too)
    for (int t = 0; t < T_SEQ; ++t) {
        // -------- layer 1: z1 = [h1_{t-1} | x_t | 1] @ W1p^T ------------------
        // Entirely register-resident: no LDS ops, no barrier dependency.
        v8f a1[4] = {vzero, vzero, vzero, vzero};   // i, f, g, o tiles
#pragma unroll
        for (int g = 0; g < 4; ++g) {
            a1[g] = __builtin_amdgcn_wmma_f32_16x16x32_f16(
                false, aA0, false, w1t[g], (short)0, a1[g], false, false);
            a1[g] = __builtin_amdgcn_wmma_f32_16x16x32_f16(
                false, aA1, false, w1t[4 + g], (short)0, a1[g], false, false);
        }
        float h1v[8];
#pragma unroll
        for (int i = 0; i < 8; ++i) {  // runs in the barrier-A shadow
            float ig = sig_hw(a1[0][i]);
            float fg = sig_hw(a1[1][i]);
            float gg = tanh_hw(a1[2][i]);
            float og = sig_hw(a1[3][i]);
            float cn = fg * c1[i] + ig * gg;
            c1[i] = cn;
            h1v[i] = og * tanh_hw(cn);
        }
        // x_{t+1} from the immutable staged tile (read-only; no hazard)
        _Float16 xn = (_Float16)0.0f;
        if (tid < MT) xn = x_s[(t + 1) * MT + tid];

        BAR_WAIT_DEP(h1v[7], h1v[3]);  // wait A: h2_{t-1} visible; all reads of
                                       // cols 0..63 (last step's partB) drained
                                       // -> cols 0..63 writable

        // -------- layer 2, h2-half: kc=2,3 read h2_{t-1} (cols 64..127) -------
        v8f a2[4] = {vzero, vzero, vzero, vzero};
#pragma unroll
        for (int kc = 2; kc < 4; ++kc) {
            v16h a = ld_tile(As, 128, 0, kc * 32, lane);
#pragma unroll
            for (int g = 0; g < 4; ++g)
                a2[g] = __builtin_amdgcn_wmma_f32_16x16x32_f16(
                    false, a, false, w2t[kc * 4 + g], (short)0, a2[g], false, false);
        }
        // publish h1_t (cols < 50) and x_{t+1} (col 50) together
        if (col < HH1) {
#pragma unroll
            for (int i = 0; i < 8; ++i)
                As[(i + 8 * hi) * 128 + col] = (_Float16)h1v[i];
        }
        if (tid < MT)
            As[tid * 128 + 50] = xn;
        BAR_SIGNAL();                  // sig B: partA reads + h1/x stores drained
                                       // (partA WMMAs fill this barrier's gap)
        BAR_WAIT();                    // wait B: h1_t, x_{t+1} visible; all
                                       // col-64..127 reads drained -> writable

        // -------- layer 2, h1-half: kc=0,1 read [h1_t | x_{t+1} | 1] ----------
        // (W2 row k=50 is zero, so the x_{t+1} column is ignored here.)
        // These loads ARE next iteration's layer-1 A fragments.
        aA0 = ld_tile(As, 128, 0, 0, lane);
        aA1 = ld_tile(As, 128, 0, 32, lane);
#pragma unroll
        for (int g = 0; g < 4; ++g) {
            a2[g] = __builtin_amdgcn_wmma_f32_16x16x32_f16(
                false, aA0, false, w2t[g], (short)0, a2[g], false, false);
            a2[g] = __builtin_amdgcn_wmma_f32_16x16x32_f16(
                false, aA1, false, w2t[4 + g], (short)0, a2[g], false, false);
        }
        float h2v[8];
#pragma unroll
        for (int i = 0; i < 8; ++i) {
            float ig = sig_hw(a2[0][i]);
            float fg = sig_hw(a2[1][i]);
            float gg = tanh_hw(a2[2][i]);
            float og = sig_hw(a2[3][i]);
            float cn = fg * c2[i] + ig * gg;
            c2[i] = cn;
            h2v[i] = og * tanh_hw(cn);
        }
        // h2 stores are safe without another barrier: every wave's reads of
        // cols 64..127 were drained before its sig B, and we are past wait B.
#pragma unroll
        for (int i = 0; i < 8; ++i)
            As[(i + 8 * hi) * 128 + 64 + col] = (_Float16)h2v[i];
        BAR_SIGNAL();                  // sig A: h2_t (+ partB reads) drained
    }
    BAR_WAIT();                        // final state visible

    // ---------------- tiny FC head on final h2 (16 rows, scalar) --------------
    if (tid < MT) {
        const int row = tid;
        float f1[32];
#pragma unroll 4
        for (int o = 0; o < 32; ++o) {
            float s = fc1_b[o];
            for (int k = 0; k < 64; ++k)
                s += (float)As[row * 128 + 64 + k] * fc1_w[o * 64 + k];
            f1[o] = s;
        }
        float f2[16];
#pragma unroll 4
        for (int o = 0; o < 16; ++o) {
            float s = fc2_b[o];
            for (int k = 0; k < 32; ++k)
                s += f1[k] * fc2_w[o * 32 + k];
            f2[o] = s;
        }
        float s = fc3_b[0];
        for (int k = 0; k < 16; ++k)
            s += f2[k] * fc3_w[k];
        out[gbase + row] = s;
    }
}

extern "C" void kernel_launch(void* const* d_in, const int* in_sizes, int n_in,
                              void* d_out, int out_size, void* d_ws, size_t ws_size,
                              hipStream_t stream) {
    const float* x     = (const float*)d_in[0];
    const float* w_ih1 = (const float*)d_in[1];
    const float* w_hh1 = (const float*)d_in[2];
    const float* b_ih1 = (const float*)d_in[3];
    const float* b_hh1 = (const float*)d_in[4];
    const float* w_ih2 = (const float*)d_in[5];
    const float* w_hh2 = (const float*)d_in[6];
    const float* b_ih2 = (const float*)d_in[7];
    const float* b_hh2 = (const float*)d_in[8];
    const float* fc1_w = (const float*)d_in[9];
    const float* fc1_b = (const float*)d_in[10];
    const float* fc2_w = (const float*)d_in[11];
    const float* fc2_b = (const float*)d_in[12];
    const float* fc3_w = (const float*)d_in[13];
    const float* fc3_b = (const float*)d_in[14];

    const int B = in_sizes[0] / T_SEQ;          // IN == 1
    const int grid = B / MT;                    // 2048/16 = 128 workgroups
    const size_t shmem =
        (size_t)(256 * 64 + 256 * 128 + MT * 128 + (T_SEQ + 1) * MT) *
        sizeof(_Float16);                       // ~117 KB
    lstm2_wmma_fused<<<grid, 128, shmem, stream>>>(
        x, w_ih1, w_hh1, b_ih1, b_hh1,
        w_ih2, w_hh2, b_ih2, b_hh2,
        fc1_w, fc1_b, fc2_w, fc2_b, fc3_w, fc3_b,
        (float*)d_out);
}
